// PointerGeneratorSeq2SeqBiDirectional_82085414961500
// MI455X (gfx1250) — matrix operand order repaired and checked by reading, not compiled, not run on hardware
//
#include <hip/hip_runtime.h>

// ---------------------------------------------------------------------------
// Pointer-generator seq2seq (bi-GRU encoder + GRU decoder + attention + copy)
// MI455X / gfx1250: all GEMMs via V_WMMA_F32_16X16X32_F16 (wave32).
// Each wave computes a 16x32 C tile, K unrolled by 2 -> 4 WMMAs per loop body.
// ---------------------------------------------------------------------------

#define VSZ  32000
#define DDIM 256
#define BSZ  4
#define SLEN 512
#define TLEN 256

typedef __attribute__((ext_vector_type(16))) _Float16 v16h;
typedef __attribute__((ext_vector_type(8)))  _Float16 v8h;
typedef __attribute__((ext_vector_type(8)))  float    v8f;

union AFrag { v16h v; v8h h[2]; };

__device__ __forceinline__ float sigmoidf_(float x) {
    return 1.0f / (1.0f + __expf(-x));
}

// ----------------------------- elementwise -------------------------------

__global__ void cvt1d_k(const float* __restrict__ s, _Float16* __restrict__ d, long n) {
    long i = (long)blockIdx.x * blockDim.x + threadIdx.x;
    if (i < n) d[i] = (_Float16)s[i];
}

__global__ void cvt2d_k(const float* __restrict__ s, int sld,
                        _Float16* __restrict__ d, int dld, int rows, int cols) {
    long i = (long)blockIdx.x * blockDim.x + threadIdx.x;
    long n = (long)rows * cols;
    if (i >= n) return;
    int r = (int)(i / cols);
    int c = (int)(i - (long)r * cols);
    d[(long)r * dld + c] = (_Float16)s[(long)r * sld + c];
}

__global__ void trans_k(const float* __restrict__ s, float* __restrict__ d, int R, int C) {
    long i = (long)blockIdx.x * blockDim.x + threadIdx.x;
    if (i >= (long)R * C) return;
    int r = (int)(i / C);
    int c = (int)(i - (long)r * C);
    d[(long)c * R + r] = s[i];
}

// per-batch f16 transpose: d[b, c, l] = s[b, l, c]  (s: [batch, L, C])
__global__ void trans16_k(const _Float16* __restrict__ s, _Float16* __restrict__ d,
                          int L, int C, long n) {
    long i = (long)blockIdx.x * blockDim.x + threadIdx.x;
    if (i >= n) return;
    long per = (long)L * C;
    int  b   = (int)(i / per);
    long rem = i - (long)b * per;
    int  l   = (int)(rem / C);
    int  c   = (int)(rem - (long)l * C);
    d[(long)b * per + (long)c * L + l] = s[i];
}

// out[b,l,:] = emb[tok[b,l],:] + pos[l,:]
__global__ void embed_k(const int* __restrict__ tok, const float* __restrict__ emb,
                        const float* __restrict__ pos, float* __restrict__ out,
                        int L, long n) {
    long i = (long)blockIdx.x * blockDim.x + threadIdx.x;
    if (i >= n) return;
    int  dcol = (int)(i & (DDIM - 1));
    long bl   = i >> 8;                 // DDIM == 256
    int  l    = (int)(bl % L);
    int  t    = tok[bl];
    out[i] = emb[(long)t * DDIM + dcol] + pos[(long)l * DDIM + dcol];
}

// ----------------------------- WMMA GEMM ---------------------------------
// C[m,n] = alpha * sum_k A[m,k]*B[n,k]  (+ bias[n])
//   A element (m,k) at A[m*lda + k]   (k contiguous, f16)
//   B element (n,k) at B[n*ldb + k]   (k contiguous, f16)
//   C element (m,n) at C[m*ldc + n]   (f32)
// One wave computes a 16x32 tile (two acc chains share the A fragment);
// K unrolled by 2. Requires M%16==0, N%32==0, K%64==0; lda,ldb multiples of 8.

__global__ __launch_bounds__(128)
void wmma_gemm_k(const _Float16* __restrict__ A, int lda, long bsA,
                 const _Float16* __restrict__ B, int ldb, long bsB,
                 float* __restrict__ C, int ldc, long bsC,
                 const float* __restrict__ bias, float alpha,
                 int K, int tilesN32) {
    int wave  = threadIdx.x >> 5;
    int lane  = threadIdx.x & 31;
    int tileN = blockIdx.x * 4 + wave;          // 32-column tiles
    if (tileN >= tilesN32) return;              // wave-uniform: EXEC stays all-ones
    int tileM = blockIdx.y;

    const _Float16* Ab = A + (long)blockIdx.z * bsA;
    const _Float16* Bb = B + (long)blockIdx.z * bsB;
    float*          Cb = C + (long)blockIdx.z * bsC;

    int m0 = tileM * 16, n0 = tileN * 32;
    int half = lane >> 4;                       // which K-half this lane holds
    int nl   = lane & 15;
    int m = m0 + nl;                            // A row owned by this lane
    int n = n0 + nl;                            // C columns n and n+16

    // ISA fragment layouts:
    //  A 16x32: lane's data = two contiguous runs of 8 at K = half*8 and 16+half*8
    //  B 32x16: lane's data = two contiguous runs of 8 at K = half*8 and 16+half*8, column = lane&15
    const _Float16* ap  = Ab + (long)m * lda + half * 8;
    const _Float16* bp0 = Bb + (long)n        * ldb + half * 8;
    const _Float16* bp1 = Bb + (long)(n + 16) * ldb + half * 8;

    v8f acc0 = {}, acc1 = {};
    for (int kb = 0; kb < K; kb += 64) {
        AFrag a0, a1, b00, b10, b01, b11;
        a0.h[0]  = *(const v8h*)(ap);       a0.h[1]  = *(const v8h*)(ap + 16);
        a1.h[0]  = *(const v8h*)(ap + 32);  a1.h[1]  = *(const v8h*)(ap + 48);
        b00.h[0] = *(const v8h*)(bp0);      b00.h[1] = *(const v8h*)(bp0 + 16);
        b10.h[0] = *(const v8h*)(bp1);      b10.h[1] = *(const v8h*)(bp1 + 16);
        b01.h[0] = *(const v8h*)(bp0 + 32); b01.h[1] = *(const v8h*)(bp0 + 48);
        b11.h[0] = *(const v8h*)(bp1 + 32); b11.h[1] = *(const v8h*)(bp1 + 48);

        acc0 = __builtin_amdgcn_wmma_f32_16x16x32_f16(false, a0.v, false, b00.v, (short)0, acc0, false, false);
        acc1 = __builtin_amdgcn_wmma_f32_16x16x32_f16(false, a0.v, false, b10.v, (short)0, acc1, false, false);
        acc0 = __builtin_amdgcn_wmma_f32_16x16x32_f16(false, a1.v, false, b01.v, (short)0, acc0, false, false);
        acc1 = __builtin_amdgcn_wmma_f32_16x16x32_f16(false, a1.v, false, b11.v, (short)0, acc1, false, false);

        ap += 64; bp0 += 64; bp1 += 64;
    }

    float bn0 = bias ? bias[n]      : 0.0f;
    float bn1 = bias ? bias[n + 16] : 0.0f;
    #pragma unroll
    for (int r = 0; r < 8; ++r) {
        int row = m0 + half * 8 + r;            // ISA C layout: VGPR r -> M = 8*half + r
        Cb[(long)row * ldc + n]      = acc0[r] * alpha + bn0;
        Cb[(long)row * ldc + n + 16] = acc1[r] * alpha + bn1;
    }
}

// ----------------------------- GRU scan ----------------------------------
// One block = one full sequential scan (B=4 batch lanes kept in LDS).
// whhT is Whh transposed: whhT[k*768 + j] = Whh[j*256 + k]  (coalesced reads).

struct GruParams {
    const float* xW;     // [B, L, 768] precomputed x@Wih^T + bih
    const float* whhT;   // [256, 768]
    const float* bhh;    // [768]
    float*       out;    // [B, L, outLd]
    int          rev;    // iterate time reversed
    int          colOff; // column offset into out rows
};

__global__ __launch_bounds__(768)
void gru_scan_k(GruParams p0, GruParams p1, int L, int outLd) {
    GruParams p = (blockIdx.x == 0) ? p0 : p1;
    __shared__ float h[BSZ][DDIM];
    __shared__ float hW[BSZ][3 * DDIM];
    int j = threadIdx.x;                       // 0..767
    if (j < DDIM)
        for (int b = 0; b < BSZ; ++b) h[b][j] = 0.0f;
    __syncthreads();

    for (int ti = 0; ti < L; ++ti) {
        int t = p.rev ? (L - 1 - ti) : ti;

        // hW = h @ Whh^T + bhh : thread j produces hW[b][j] for all b
        float a0 = p.bhh[j], a1 = a0, a2 = a0, a3 = a0;
        for (int k = 0; k < DDIM; ++k) {
            float w = p.whhT[k * (3 * DDIM) + j];
            a0 = fmaf(h[0][k], w, a0);
            a1 = fmaf(h[1][k], w, a1);
            a2 = fmaf(h[2][k], w, a2);
            a3 = fmaf(h[3][k], w, a3);
        }
        hW[0][j] = a0; hW[1][j] = a1; hW[2][j] = a2; hW[3][j] = a3;
        __syncthreads();

        if (j < DDIM) {
            for (int b = 0; b < BSZ; ++b) {
                const float* xw = p.xW + ((long)b * L + t) * (3 * DDIM);
                float r  = sigmoidf_(xw[j]            + hW[b][j]);
                float z  = sigmoidf_(xw[DDIM + j]     + hW[b][DDIM + j]);
                float nn = tanhf    (xw[2 * DDIM + j] + r * hW[b][2 * DDIM + j]);
                float hn = (1.0f - z) * nn + z * h[b][j];
                h[b][j] = hn;
                p.out[((long)b * L + t) * outLd + p.colOff + j] = hn;
            }
        }
        __syncthreads();
    }
}

// ----------------------------- softmax (scores -> probs) -----------------
// One block per (b,t) row of length S. Writes f32 in place + f16 copy.

__global__ __launch_bounds__(256)
void softmax_k(float* __restrict__ sc, _Float16* __restrict__ p16, int S) {
    __shared__ float red[256];
    float* row = sc + (long)blockIdx.x * S;
    int tid = threadIdx.x;

    float m = -3.4e38f;
    for (int s = tid; s < S; s += blockDim.x) m = fmaxf(m, row[s]);
    red[tid] = m; __syncthreads();
    for (int w = 128; w > 0; w >>= 1) {
        if (tid < w) red[tid] = fmaxf(red[tid], red[tid + w]);
        __syncthreads();
    }
    m = red[0]; __syncthreads();

    float sum = 0.0f;
    for (int s = tid; s < S; s += blockDim.x) sum += __expf(row[s] - m);
    red[tid] = sum; __syncthreads();
    for (int w = 128; w > 0; w >>= 1) {
        if (tid < w) red[tid] += red[tid + w];
        __syncthreads();
    }
    float inv = 1.0f / red[0];

    for (int s = tid; s < S; s += blockDim.x) {
        float p = __expf(row[s] - m) * inv;
        row[s] = p;
        p16[(long)blockIdx.x * S + s] = (_Float16)p;
    }
}

// ----------------------------- gate --------------------------------------
// gate[r] = sigmoid( dec[r,:].Wg[0:256] + ctx[r,:].Wg[256:768] + bg )

__global__ __launch_bounds__(256)
void gate_k(const float* __restrict__ dec, const float* __restrict__ ctx,
            const float* __restrict__ Wg, const float* __restrict__ bg,
            float* __restrict__ gate, int rows) {
    int wave = (blockIdx.x * blockDim.x + threadIdx.x) >> 5;
    int lane = threadIdx.x & 31;
    if (wave >= rows) return;
    float s = 0.0f;
    for (int k = lane; k < DDIM; k += 32)     s = fmaf(dec[(long)wave * DDIM + k], Wg[k], s);
    for (int k = lane; k < 2 * DDIM; k += 32) s = fmaf(ctx[(long)wave * 2 * DDIM + k], Wg[DDIM + k], s);
    #pragma unroll
    for (int off = 16; off > 0; off >>= 1) s += __shfl_down(s, off, 32);
    if (lane == 0) gate[wave] = sigmoidf_(s + bg[0]);
}

// ----------------------------- softmax+gate+copy blend --------------------
// In-place over logits rows of d_out: out = g*softmax(logits) + (1-g)*P_copy.
// Copy-scatter done by one thread per row for deterministic accumulation.

__global__ __launch_bounds__(256)
void blend_k(float* __restrict__ out, const float* __restrict__ probs,
             const float* __restrict__ gate, const int* __restrict__ src) {
    __shared__ float red[256];
    int row = blockIdx.x;                      // b*T + t
    int b   = row / TLEN;
    float* o = out + (long)row * VSZ;
    int tid = threadIdx.x;

    float m = -3.4e38f;
    for (int v = tid; v < VSZ; v += blockDim.x) m = fmaxf(m, o[v]);
    red[tid] = m; __syncthreads();
    for (int w = 128; w > 0; w >>= 1) {
        if (tid < w) red[tid] = fmaxf(red[tid], red[tid + w]);
        __syncthreads();
    }
    m = red[0]; __syncthreads();

    float sum = 0.0f;
    for (int v = tid; v < VSZ; v += blockDim.x) sum += __expf(o[v] - m);
    red[tid] = sum; __syncthreads();
    for (int w = 128; w > 0; w >>= 1) {
        if (tid < w) red[tid] += red[tid + w];
        __syncthreads();
    }
    float g   = gate[row];
    float inv = g / red[0];

    for (int v = tid; v < VSZ; v += blockDim.x)
        o[v] = __expf(o[v] - m) * inv;
    __threadfence();
    __syncthreads();

    if (tid == 0) {
        float og = 1.0f - g;
        for (int s = 0; s < SLEN; ++s) {
            int tok = src[(long)b * SLEN + s];
            o[tok] += og * probs[(long)row * SLEN + s];
        }
    }
}

// ---------------------------------------------------------------------------

extern "C" void kernel_launch(void* const* d_in, const int* in_sizes, int n_in,
                              void* d_out, int out_size, void* d_ws, size_t ws_size,
                              hipStream_t stream) {
    (void)in_sizes; (void)n_in; (void)out_size; (void)ws_size;

    const int*   src    = (const int*)  d_in[0];
    const int*   tgt    = (const int*)  d_in[1];
    const float* emb    = (const float*)d_in[2];
    const float* pos    = (const float*)d_in[3];
    const float* ef_wih = (const float*)d_in[4];
    const float* ef_whh = (const float*)d_in[5];
    const float* ef_bih = (const float*)d_in[6];
    const float* ef_bhh = (const float*)d_in[7];
    const float* eb_wih = (const float*)d_in[8];
    const float* eb_whh = (const float*)d_in[9];
    const float* eb_bih = (const float*)d_in[10];
    const float* eb_bhh = (const float*)d_in[11];
    const float* d_wih  = (const float*)d_in[12];
    const float* d_whh  = (const float*)d_in[13];
    const float* d_bih  = (const float*)d_in[14];
    const float* d_bhh  = (const float*)d_in[15];
    const float* Wq     = (const float*)d_in[16];
    const float* Wk     = (const float*)d_in[17];
    const float* Wgen   = (const float*)d_in[18];
    const float* bgen   = (const float*)d_in[19];
    const float* Wgate  = (const float*)d_in[20];
    const float* bgate  = (const float*)d_in[21];
    float* out = (float*)d_out;

    // ---- workspace carve-out (256B aligned blocks) ----
    char*  w   = (char*)d_ws;
    size_t off = 0;
    auto alloc = [&](size_t bytes) -> void* {
        off = (off + 255) & ~(size_t)255;
        void* p = w + off;
        off += bytes;
        return p;
    };
    const long ML3 = 3 * DDIM;                         // 768
    float* srcEmb  = (float*)alloc((long)BSZ*SLEN*DDIM * 4);
    float* tgtEmb  = (float*)alloc((long)BSZ*TLEN*DDIM * 4);
    float* xW_ef   = (float*)alloc((long)BSZ*SLEN*ML3  * 4);
    float* xW_eb   = (float*)alloc((long)BSZ*SLEN*ML3  * 4);
    float* xW_d    = (float*)alloc((long)BSZ*TLEN*ML3  * 4);
    float* whhT_ef = (float*)alloc((long)ML3*DDIM      * 4);
    float* whhT_eb = (float*)alloc((long)ML3*DDIM      * 4);
    float* whhT_d  = (float*)alloc((long)ML3*DDIM      * 4);
    float* encOut  = (float*)alloc((long)BSZ*SLEN*2*DDIM * 4);
    float* decOut  = (float*)alloc((long)BSZ*TLEN*DDIM * 4);
    float* Qbuf    = (float*)alloc((long)BSZ*TLEN*DDIM * 4);
    float* Kbuf    = (float*)alloc((long)BSZ*SLEN*DDIM * 4);
    float* scores  = (float*)alloc((long)BSZ*TLEN*SLEN * 4);   // becomes probs
    float* ctx     = (float*)alloc((long)BSZ*TLEN*2*DDIM * 4);
    float* gate    = (float*)alloc((long)BSZ*TLEN * 4);
    _Float16* srcEmb16 = (_Float16*)alloc((long)BSZ*SLEN*DDIM * 2);
    _Float16* tgtEmb16 = (_Float16*)alloc((long)BSZ*TLEN*DDIM * 2);
    _Float16* wih16_ef = (_Float16*)alloc((long)ML3*DDIM * 2);
    _Float16* wih16_eb = (_Float16*)alloc((long)ML3*DDIM * 2);
    _Float16* wih16_d  = (_Float16*)alloc((long)ML3*DDIM * 2);
    _Float16* Wq16     = (_Float16*)alloc((long)DDIM*DDIM * 2);
    _Float16* Wk16     = (_Float16*)alloc((long)DDIM*2*DDIM * 2);
    _Float16* Wgen16   = (_Float16*)alloc((long)VSZ*ML3 * 2);
    _Float16* enc16    = (_Float16*)alloc((long)BSZ*SLEN*2*DDIM * 2);
    _Float16* encT16   = (_Float16*)alloc((long)BSZ*SLEN*2*DDIM * 2); // [b][2D][S]
    _Float16* Q16      = (_Float16*)alloc((long)BSZ*TLEN*DDIM * 2);
    _Float16* K16      = (_Float16*)alloc((long)BSZ*SLEN*DDIM * 2);
    _Float16* probs16  = (_Float16*)alloc((long)BSZ*TLEN*SLEN * 2);
    _Float16* comb16   = (_Float16*)alloc((long)BSZ*TLEN*ML3 * 2);

    auto cvt = [&](const float* s, _Float16* d, long n) {
        cvt1d_k<<<dim3((unsigned)((n + 255) / 256)), 256, 0, stream>>>(s, d, n);
    };

    // 1) weight conversions + Whh transposes
    cvt(ef_wih, wih16_ef, (long)ML3*DDIM);
    cvt(eb_wih, wih16_eb, (long)ML3*DDIM);
    cvt(d_wih,  wih16_d,  (long)ML3*DDIM);
    cvt(Wq,     Wq16,     (long)DDIM*DDIM);
    cvt(Wk,     Wk16,     (long)DDIM*2*DDIM);
    cvt(Wgen,   Wgen16,   (long)VSZ*ML3);
    {
        unsigned g = (unsigned)(((long)ML3*DDIM + 255) / 256);
        trans_k<<<g, 256, 0, stream>>>(ef_whh, whhT_ef, ML3, DDIM);
        trans_k<<<g, 256, 0, stream>>>(eb_whh, whhT_eb, ML3, DDIM);
        trans_k<<<g, 256, 0, stream>>>(d_whh,  whhT_d,  ML3, DDIM);
    }

    // 2) embeddings (+f16 copies)
    {
        long ns = (long)BSZ*SLEN*DDIM, nt = (long)BSZ*TLEN*DDIM;
        embed_k<<<dim3((unsigned)((ns + 255) / 256)), 256, 0, stream>>>(src, emb, pos, srcEmb, SLEN, ns);
        embed_k<<<dim3((unsigned)((nt + 255) / 256)), 256, 0, stream>>>(tgt, emb, pos, tgtEmb, TLEN, nt);
        cvt(srcEmb, srcEmb16, ns);
        cvt(tgtEmb, tgtEmb16, nt);
    }

    // 3) GRU input projections: xW = x @ Wih^T + bih   (M x 768, K=256)
    wmma_gemm_k<<<dim3(6, (BSZ*SLEN)/16, 1), 128, 0, stream>>>(
        srcEmb16, DDIM, 0, wih16_ef, DDIM, 0, xW_ef, ML3, 0, ef_bih, 1.0f, DDIM, (int)ML3/32);
    wmma_gemm_k<<<dim3(6, (BSZ*SLEN)/16, 1), 128, 0, stream>>>(
        srcEmb16, DDIM, 0, wih16_eb, DDIM, 0, xW_eb, ML3, 0, eb_bih, 1.0f, DDIM, (int)ML3/32);
    wmma_gemm_k<<<dim3(6, (BSZ*TLEN)/16, 1), 128, 0, stream>>>(
        tgtEmb16, DDIM, 0, wih16_d, DDIM, 0, xW_d, ML3, 0, d_bih, 1.0f, DDIM, (int)ML3/32);

    // 4) GRU scans: encoder fwd+bwd concurrently (2 blocks), then decoder
    GruParams pf{xW_ef, whhT_ef, ef_bhh, encOut, 0, 0};
    GruParams pb{xW_eb, whhT_eb, eb_bhh, encOut, 1, DDIM};
    gru_scan_k<<<2, 768, 0, stream>>>(pf, pb, SLEN, 2*DDIM);
    GruParams pd{xW_d, whhT_d, d_bhh, decOut, 0, 0};
    gru_scan_k<<<1, 768, 0, stream>>>(pd, pd, TLEN, DDIM);

    // 5) f16 copies of GRU outputs; dec goes into combined cols [0:256);
    //    enc16 also transposed per batch -> encT16[b][d][s] for the ctx GEMMs
    cvt(encOut, enc16, (long)BSZ*SLEN*2*DDIM);
    {
        long n = (long)BSZ*TLEN*DDIM;
        cvt2d_k<<<dim3((unsigned)((n + 255) / 256)), 256, 0, stream>>>(
            decOut, DDIM, comb16, ML3, BSZ*TLEN, DDIM);
        long ne = (long)BSZ*SLEN*2*DDIM;
        trans16_k<<<dim3((unsigned)((ne + 255) / 256)), 256, 0, stream>>>(
            enc16, encT16, SLEN, 2*DDIM, ne);
    }

    // 6) Q = dec @ Wq^T ; K = enc @ Wk^T
    wmma_gemm_k<<<dim3(2, (BSZ*TLEN)/16, 1), 128, 0, stream>>>(
        comb16, ML3, 0, Wq16, DDIM, 0, Qbuf, DDIM, 0, nullptr, 1.0f, DDIM, DDIM/32);
    wmma_gemm_k<<<dim3(2, (BSZ*SLEN)/16, 1), 128, 0, stream>>>(
        enc16, 2*DDIM, 0, Wk16, 2*DDIM, 0, Kbuf, DDIM, 0, nullptr, 1.0f, 2*DDIM, DDIM/32);
    cvt(Qbuf, Q16, (long)BSZ*TLEN*DDIM);
    cvt(Kbuf, K16, (long)BSZ*SLEN*DDIM);

    // 7) scores = Q K^T / 16  (batched over B), softmax over S
    wmma_gemm_k<<<dim3(4, TLEN/16, BSZ), 128, 0, stream>>>(
        Q16, DDIM, (long)TLEN*DDIM, K16, DDIM, (long)SLEN*DDIM,
        scores, SLEN, (long)TLEN*SLEN, nullptr, 1.0f/16.0f, DDIM, SLEN/32);
    softmax_k<<<BSZ*TLEN, 256, 0, stream>>>(scores, probs16, SLEN);

    // 8) ctx_f / ctx_b = probs @ enc slices (fast path via encT16)
    wmma_gemm_k<<<dim3(2, TLEN/16, BSZ), 128, 0, stream>>>(
        probs16, SLEN, (long)TLEN*SLEN, encT16, SLEN, (long)SLEN*2*DDIM,
        ctx, 2*DDIM, (long)TLEN*2*DDIM, nullptr, 1.0f, SLEN, DDIM/32);
    wmma_gemm_k<<<dim3(2, TLEN/16, BSZ), 128, 0, stream>>>(
        probs16, SLEN, (long)TLEN*SLEN, encT16 + (long)DDIM*SLEN, SLEN, (long)SLEN*2*DDIM,
        ctx + DDIM, 2*DDIM, (long)TLEN*2*DDIM, nullptr, 1.0f, SLEN, DDIM/32);

    // 9) gate; assemble combined f16 cols [256:768)
    gate_k<<<(BSZ*TLEN*32 + 255)/256, 256, 0, stream>>>(decOut, ctx, Wgate, bgate, gate, BSZ*TLEN);
    {
        long n = (long)BSZ*TLEN*2*DDIM;
        cvt2d_k<<<dim3((unsigned)((n + 255) / 256)), 256, 0, stream>>>(
            ctx, 2*DDIM, comb16 + DDIM, ML3, BSZ*TLEN, 2*DDIM);
    }

    // 10) logits = combined @ Wgen^T + bgen  -> d_out  (the 50-GFLOP GEMM)
    wmma_gemm_k<<<dim3(VSZ/32/4, (BSZ*TLEN)/16, 1), 128, 0, stream>>>(
        comb16, ML3, 0, Wgen16, ML3, 0, out, VSZ, 0, bgen, 1.0f, ML3, VSZ/32);

    // 11) in-place: out = gate*softmax(logits) + (1-gate)*P_copy
    blend_k<<<BSZ*TLEN, 256, 0, stream>>>(out, scores, gate, src);
}
